// MultiScaleGNN_20306605375937
// MI455X (gfx1250) — matrix operand order repaired and checked by reading, not compiled, not run on hardware
//
#include <hip/hip_runtime.h>
#include <hip/hip_bf16.h>

// MultiScaleGNN on MI455X (gfx1250): scatter/gather via f32 atomics (L2-resident
// feature rows), all dense GEMMs via v_wmma_f32_16x16x32_bf16 with fused
// 1/deg scaling, bias, LayerNorm+ReLU epilogues. wave32, 8 waves/block,
// 32x128 output tile per block (2 M-tiles per wave -> 4 WMMAs per K-step,
// B fragments reused across both M-tiles).

typedef __attribute__((ext_vector_type(16))) __bf16 v16bf;
typedef __attribute__((ext_vector_type(8)))  float  v8f;

#define HIDDEN 128
#define MTILE  32
#define LN_EPS 1e-5f

union AFrag { v16bf v; unsigned int u[8]; };

__device__ __forceinline__ unsigned short f2bf(float f) {
    unsigned int u = __float_as_uint(f);
    unsigned int r = u + 0x7FFFu + ((u >> 16) & 1u);   // round-to-nearest-even
    return (unsigned short)(r >> 16);
}

// ---------------- graph aggregation kernels ----------------

__global__ void count_deg_kernel(const long long* __restrict__ dst,
                                 float* __restrict__ cnt, int E) {
    int e = blockIdx.x * blockDim.x + threadIdx.x;
    if (e < E) atomicAdd(&cnt[dst[e]], 1.0f);
}

__global__ void inv_cnt_kernel(const float* __restrict__ cnt,
                               float* __restrict__ invc, int N) {
    int i = blockIdx.x * blockDim.x + threadIdx.x;
    if (i < N) invc[i] = 1.0f / fmaxf(cnt[i], 1.0f);
}

// one thread per (edge, 4-feature chunk): gather float4 from src row, atomic-add
// into dst row. Feature matrices are L2-resident (192MB L2), so random rows hit L2.
__global__ void scatter_add_kernel(const float* __restrict__ h,
                                   const long long* __restrict__ src,
                                   const long long* __restrict__ dst,
                                   float* __restrict__ agg, int E, int W4) {
    long long tid = (long long)blockIdx.x * blockDim.x + threadIdx.x;
    long long total = (long long)E * W4;
    if (tid >= total) return;
    int e  = (int)(tid / W4);
    int f4 = (int)(tid % W4);
    long long s = src[e], d = dst[e];
    int W = W4 * 4;
    const float4 v = *(const float4*)(h + s * W + (long long)f4 * 4);
    float* p = agg + d * W + (long long)f4 * 4;
    atomicAdd(p + 0, v.x); atomicAdd(p + 1, v.y);
    atomicAdd(p + 2, v.z); atomicAdd(p + 3, v.w);
}

// ---------------- weight prep: f32 (K x 128) row-major -> bf16 transposed (128 x K) ----------------

__global__ void prep_weight_kernel(const float* __restrict__ src,
                                   unsigned short* __restrict__ dstT,
                                   int K, int rowoff) {
    int tid = blockIdx.x * blockDim.x + threadIdx.x;
    if (tid >= K * HIDDEN) return;
    int k = tid / HIDDEN, n = tid % HIDDEN;
    dstT[(size_t)n * K + k] = f2bf(src[(size_t)(k + rowoff) * HIDDEN + n]);
}

// ---------------- dual-GEMM tile core (shared by conv & JK kernels) ----------------
// Computes Ct[32][128] = (A1*diag(invc)) @ W1 + A2 @ W2 + bias for rows [m0, m0+32).
// 8 waves; each wave owns one 16-wide N-tile and TWO 16-row M-tiles, so the
// B (weight) fragments are loaded once per K-step and reused by 2 WMMAs each.
__device__ __forceinline__ void gemm_dual_tile(
    const float* __restrict__ A1, const float* __restrict__ invc,
    const float* __restrict__ A2,
    const unsigned short* __restrict__ W1T, const unsigned short* __restrict__ W2T,
    const float* __restrict__ bias, int K, int m0,
    float Ct[MTILE][HIDDEN + 1],
    unsigned short As1[MTILE][36], unsigned short As2[MTILE][36])
{
    const int tid  = threadIdx.x;
    const int wave = tid >> 5;
    const int lane = tid & 31;
    const int hi   = lane >> 4;       // half-wave selector
    const int col  = lane & 15;
    const int ng   = wave * 16 + col; // global output column

    v8f c0 = {};   // rows [0,16) of the tile
    v8f c1 = {};   // rows [16,32)
    for (int k0 = 0; k0 < K; k0 += 32) {
        __syncthreads();
        // stage 32x32 A-chunks into LDS as bf16 (1/deg scaling fused for A1)
        for (int idx = tid; idx < MTILE * 32; idx += 256) {
            int r = idx >> 5, kk = idx & 31;
            long long grow = m0 + r;
            float sc = invc ? invc[grow] : 1.0f;
            As1[r][kk] = f2bf(A1[grow * K + k0 + kk] * sc);
            As2[r][kk] = f2bf(A2[grow * K + k0 + kk]);
        }
        __syncthreads();

        AFrag a1lo, a2lo, a1hi, a2hi, b1, b2;
        // B fragment: W^T[n][k], K pairs contiguous -> 8 consecutive dwords
        const unsigned int* w1p = (const unsigned int*)(W1T + (size_t)ng * K + k0 + 16 * hi);
        const unsigned int* w2p = (const unsigned int*)(W2T + (size_t)ng * K + k0 + 16 * hi);
#pragma unroll
        for (int v = 0; v < 8; ++v) {
            // A packing here (16-bit, 16x32; contiguous k, same as B): VGPR v -> K = 2v + 16*hi
            int kb = 2 * v + 16 * hi;
            a1lo.u[v] = *(const unsigned int*)&As1[col][kb];
            a2lo.u[v] = *(const unsigned int*)&As2[col][kb];
            a1hi.u[v] = *(const unsigned int*)&As1[16 + col][kb];
            a2hi.u[v] = *(const unsigned int*)&As2[16 + col][kb];
            b1.u[v] = w1p[v];
            b2.u[v] = w2p[v];
        }
        c0 = __builtin_amdgcn_wmma_f32_16x16x32_bf16(false, a1lo.v, false, b1.v,
                                                     (short)0, c0, false, false);
        c1 = __builtin_amdgcn_wmma_f32_16x16x32_bf16(false, a1hi.v, false, b1.v,
                                                     (short)0, c1, false, false);
        c0 = __builtin_amdgcn_wmma_f32_16x16x32_bf16(false, a2lo.v, false, b2.v,
                                                     (short)0, c0, false, false);
        c1 = __builtin_amdgcn_wmma_f32_16x16x32_bf16(false, a2hi.v, false, b2.v,
                                                     (short)0, c1, false, false);
    }
    __syncthreads();
    float bb = bias[ng];
#pragma unroll
    for (int r = 0; r < 8; ++r) {           // C layout: VGPR r -> M = r + 8*hi
        Ct[r + 8 * hi][ng]      = c0[r] + bb;
        Ct[16 + r + 8 * hi][ng] = c1[r] + bb;
    }
    __syncthreads();
}

// ---------------- SAGE conv + LayerNorm + ReLU ----------------

__global__ void __launch_bounds__(256)
conv_ln_relu_kernel(const float* __restrict__ A1, const float* __restrict__ invc,
                    const float* __restrict__ A2,
                    const unsigned short* __restrict__ W1T,
                    const unsigned short* __restrict__ W2T,
                    const float* __restrict__ bias,
                    const float* __restrict__ gamma, const float* __restrict__ beta,
                    float* __restrict__ out, int K)
{
    __shared__ float Ct[MTILE][HIDDEN + 1];
    __shared__ unsigned short As1[MTILE][36];
    __shared__ unsigned short As2[MTILE][36];
    int m0 = blockIdx.x * MTILE;
    gemm_dual_tile(A1, invc, A2, W1T, W2T, bias, K, m0, Ct, As1, As2);

    // fused LayerNorm + ReLU: block holds full rows, one thread per row
    if (threadIdx.x < MTILE) {
        int r = threadIdx.x;
        float s = 0.f, s2 = 0.f;
        for (int j = 0; j < HIDDEN; ++j) { float v = Ct[r][j]; s += v; s2 += v * v; }
        float mu  = s * (1.0f / HIDDEN);
        float var = s2 * (1.0f / HIDDEN) - mu * mu;
        float rs  = rsqrtf(var + LN_EPS);
        float* op = out + (long long)(m0 + r) * HIDDEN;
        for (int j = 0; j < HIDDEN; ++j) {
            float v = (Ct[r][j] - mu) * rs * gamma[j] + beta[j];
            op[j] = fmaxf(v, 0.0f);
        }
    }
}

// ---------------- JK concat-project + MLP head, fully fused ----------------

__global__ void __launch_bounds__(256)
jk_head_kernel(const float* __restrict__ h0, const float* __restrict__ h1,
               const unsigned short* __restrict__ WjkTa,
               const unsigned short* __restrict__ WjkTb,
               const float* __restrict__ bjk,
               const float* __restrict__ Wh1, const float* __restrict__ bh1,
               const float* __restrict__ Wh2, const float* __restrict__ bh2,
               float* __restrict__ out)
{
    __shared__ float Ct[MTILE][HIDDEN + 1];
    __shared__ unsigned short As1[MTILE][36];
    __shared__ unsigned short As2[MTILE][36];
    __shared__ float Hd[MTILE][33];
    int m0 = blockIdx.x * MTILE;
    // feats = h0 @ Wjk_top + h1 @ Wjk_bot + b_jk   (== concat([h0,h1]) @ W_jk + b)
    gemm_dual_tile(h0, nullptr, h1, WjkTa, WjkTb, bjk, HIDDEN, m0, Ct, As1, As2);

    // hidden = relu(feats @ W_h1 + b_h1): 32x32 outputs (weights L2-resident)
    for (int o = threadIdx.x; o < MTILE * 32; o += 256) {
        int r = o >> 5, j = o & 31;
        float s = bh1[j];
        for (int k = 0; k < HIDDEN; ++k) s += Ct[r][k] * Wh1[k * 32 + j];
        Hd[r][j] = fmaxf(s, 0.0f);
    }
    __syncthreads();
    // out = hidden @ W_h2 + b_h2: 32x16 outputs
    for (int o = threadIdx.x; o < MTILE * 16; o += 256) {
        int r = o >> 4, j = o & 15;
        float s = bh2[j];
        for (int k = 0; k < 32; ++k) s += Hd[r][k] * Wh2[k * 16 + j];
        out[(long long)(m0 + r) * 16 + j] = s;
    }
}

// ---------------- launcher ----------------

extern "C" void kernel_launch(void* const* d_in, const int* in_sizes, int n_in,
                              void* d_out, int out_size, void* d_ws, size_t ws_size,
                              hipStream_t stream) {
    const float*     x   = (const float*)d_in[0];
    const long long* ei  = (const long long*)d_in[1];
    const float* W_nl0 = (const float*)d_in[2];
    const float* b_nl0 = (const float*)d_in[3];
    const float* W_r0  = (const float*)d_in[4];
    const float* W_nl1 = (const float*)d_in[5];
    const float* b_nl1 = (const float*)d_in[6];
    const float* W_r1  = (const float*)d_in[7];
    const float* g0    = (const float*)d_in[8];
    const float* be0   = (const float*)d_in[9];
    const float* g1    = (const float*)d_in[10];
    const float* be1   = (const float*)d_in[11];
    const float* W_jk  = (const float*)d_in[12];
    const float* b_jk  = (const float*)d_in[13];
    const float* W_h1  = (const float*)d_in[14];
    const float* b_h1  = (const float*)d_in[15];
    const float* W_h2  = (const float*)d_in[16];
    const float* b_h2  = (const float*)d_in[17];
    float* out = (float*)d_out;

    const int N = in_sizes[0] / 256;   // 100000
    const int E = in_sizes[1] / 2;     // 1600000
    const long long* src = ei;
    const long long* dst = ei + E;

    // workspace carving (256B aligned)
    char* ws = (char*)d_ws;
    size_t off = 0;
    auto carve = [&](size_t bytes) -> char* {
        char* p = ws + off;
        off = (off + bytes + 255) & ~(size_t)255;
        return p;
    };
    float* cnt   = (float*)carve((size_t)N * 4);
    float* invc  = (float*)carve((size_t)N * 4);
    float* agg   = (float*)carve((size_t)N * 256 * 4);  // reused: layer0 (256w) then layer1 (128w)
    float* h0    = (float*)carve((size_t)N * HIDDEN * 4);
    float* h1    = (float*)carve((size_t)N * HIDDEN * 4);
    unsigned short* W1T0  = (unsigned short*)carve((size_t)HIDDEN * 256 * 2);
    unsigned short* W2T0  = (unsigned short*)carve((size_t)HIDDEN * 256 * 2);
    unsigned short* W1T1  = (unsigned short*)carve((size_t)HIDDEN * HIDDEN * 2);
    unsigned short* W2T1  = (unsigned short*)carve((size_t)HIDDEN * HIDDEN * 2);
    unsigned short* WjkTa = (unsigned short*)carve((size_t)HIDDEN * HIDDEN * 2);
    unsigned short* WjkTb = (unsigned short*)carve((size_t)HIDDEN * HIDDEN * 2);

    // weight prep (independent of graph work)
    prep_weight_kernel<<<(256 * HIDDEN + 255) / 256, 256, 0, stream>>>(W_nl0, W1T0, 256, 0);
    prep_weight_kernel<<<(256 * HIDDEN + 255) / 256, 256, 0, stream>>>(W_r0,  W2T0, 256, 0);
    prep_weight_kernel<<<(128 * HIDDEN + 255) / 256, 256, 0, stream>>>(W_nl1, W1T1, 128, 0);
    prep_weight_kernel<<<(128 * HIDDEN + 255) / 256, 256, 0, stream>>>(W_r1,  W2T1, 128, 0);
    prep_weight_kernel<<<(128 * HIDDEN + 255) / 256, 256, 0, stream>>>(W_jk,  WjkTa, 128, 0);
    prep_weight_kernel<<<(128 * HIDDEN + 255) / 256, 256, 0, stream>>>(W_jk,  WjkTb, 128, 128);

    // degree counts
    hipMemsetAsync(cnt, 0, (size_t)N * 4, stream);
    count_deg_kernel<<<(E + 255) / 256, 256, 0, stream>>>(dst, cnt, E);
    inv_cnt_kernel<<<(N + 255) / 256, 256, 0, stream>>>(cnt, invc, N);

    // ---- layer 0: scatter x -> agg(256w), conv+LN+ReLU -> h0 ----
    hipMemsetAsync(agg, 0, (size_t)N * 256 * 4, stream);
    {
        long long total = (long long)E * 64;
        scatter_add_kernel<<<(unsigned)((total + 255) / 256), 256, 0, stream>>>(
            x, src, dst, agg, E, 64);
    }
    conv_ln_relu_kernel<<<N / MTILE, 256, 0, stream>>>(
        agg, invc, x, W1T0, W2T0, b_nl0, g0, be0, h0, 256);

    // ---- layer 1: scatter h0 -> agg(128w), conv+LN+ReLU -> h1 ----
    hipMemsetAsync(agg, 0, (size_t)N * HIDDEN * 4, stream);
    {
        long long total = (long long)E * 32;
        scatter_add_kernel<<<(unsigned)((total + 255) / 256), 256, 0, stream>>>(
            h0, src, dst, agg, E, 32);
    }
    conv_ln_relu_kernel<<<N / MTILE, 256, 0, stream>>>(
        agg, invc, h0, W1T1, W2T1, b_nl1, g1, be1, h1, 128);

    // ---- JK concat projection + MLP head (fused) -> out ----
    jk_head_kernel<<<N / MTILE, 256, 0, stream>>>(
        h0, h1, WjkTa, WjkTb, b_jk, W_h1, b_h1, W_h2, b_h2, out);

    (void)n_in; (void)out_size; (void)ws_size;
}
